// Decoder_34617436405870
// MI455X (gfx1250) — matrix operand order repaired and checked by reading, not compiled, not run on hardware
//
#include <hip/hip_runtime.h>
#include <hip/hip_bf16.h>

typedef __attribute__((ext_vector_type(16))) __bf16 v16bf;
typedef __attribute__((ext_vector_type(8)))  float  v8f;
typedef __attribute__((ext_vector_type(4)))  unsigned int u32x4;
typedef __attribute__((ext_vector_type(8)))  unsigned int u32x8;

#define B_  4096
#define T_  5
#define V_  8192
#define E_  32
#define H_  512
#define O_  1024
#define H3_ 1536
#define CK_ 256          // K-chunk staged in LDS by the TDM for the embed GEMM
#define LN_EPS 1e-5f

static __device__ __forceinline__ float sigmoidf_(float x) {
    return 1.0f / (1.0f + __expf(-x));
}

// xor-butterfly add via ds_swizzle (group-of-32 xor mode, and_mask=0x1f)
#define SWZ_ADD(v, off) \
    (v) += __int_as_float(__builtin_amdgcn_ds_swizzle(__float_as_int(v), (off)))

static __device__ __forceinline__ void wait_tensorcnt0() {
#if __has_builtin(__builtin_amdgcn_s_wait_tensorcnt)
    __builtin_amdgcn_s_wait_tensorcnt(0);
#else
    asm volatile("s_wait_tensorcnt 0x0" ::: "memory");
#endif
}

// ---- WMMA fragment loaders -------------------------------------------------
// A-matrix 16x32 bf16 (wave32): lanes 0-15 row M=lane, K=0..15;
// lanes 16-31 row M=lane-16, K=16..31.
static __device__ __forceinline__ v16bf load_a_from_f32(const float* __restrict__ row,
                                                        int k0, int lane) {
    const int kb = (lane >> 4) << 4;
    v16bf a;
#pragma unroll
    for (int j = 0; j < 8; ++j) {
        a[j]     = (__bf16)row[k0 + kb + j];
        a[8 + j] = (__bf16)row[k0 + kb + 8 + j];
    }
    return a;
}

static __device__ __forceinline__ v16bf load_a_from_bf16(const __bf16* row, int k0, int lane) {
    const int kb = (lane >> 4) << 4;
    v16bf a;
#pragma unroll
    for (int j = 0; j < 8; ++j) {
        a[j]     = row[k0 + kb + j];
        a[8 + j] = row[k0 + kb + 8 + j];
    }
    return a;
}

// B-matrix 32x16 bf16: lane holds column N=(l&15); lanes 0-15 K=0..15,
// lanes 16-31 K=16..31. Panel stored column-major ([n][k], ld = ldk):
// each lane reads 16 contiguous bf16.
static __device__ __forceinline__ v16bf load_b_panel(const __bf16* P, size_t ldk,
                                                     int ncol, int k0, int lane) {
    const __bf16* p = P + (size_t)ncol * ldk + k0 + ((lane >> 4) << 4);
    v16bf b;
#pragma unroll
    for (int e = 0; e < 16; ++e) b[e] = p[e];
    return b;
}

#define WMMA_BF16(A, Bm, C) \
    __builtin_amdgcn_wmma_f32_16x16x32_bf16(false, (A), false, (Bm), (short)0, (C), false, false)

// ---- TDM: async DMA of a [32 cols x CK_] col-major bf16 panel into LDS -----
// D# group0: count=1, lds_addr, global_addr, type=2. group1: data_size=2B,
// tensor_dim0=V_, tensor_dim1=32, tile_dim0=CK_, tile_dim1=32, dim0_stride=V_.
static __device__ __forceinline__ void tdm_load_panel(__bf16* ldsDst, const __bf16* gsrc) {
    unsigned long long ga = (unsigned long long)(uintptr_t)gsrc;
    unsigned lds = (unsigned)(uintptr_t)ldsDst;   // low 32 bits of flat = LDS byte addr
    u32x4 g0;
    g0[0] = 1u;                                           // count=1 (valid descriptor)
    g0[1] = lds;                                          // lds_addr
    g0[2] = (unsigned)ga;                                 // global_addr[31:0]
    g0[3] = (unsigned)((ga >> 32) & 0x1FFFFFFu) | (2u << 30);  // addr[56:32] | type=2
    u32x8 g1;
    g1[0] = (1u << 16);                                   // data_size=1 -> 2 bytes
    g1[1] = ((unsigned)V_ & 0xffffu) << 16;               // tensor_dim0[15:0] @ bits63:48
    g1[2] = (((unsigned)V_ >> 16) & 0xffffu) | (32u << 16);  // dim0 hi | tensor_dim1 lo
    g1[3] = ((unsigned)CK_ << 16);                        // tensor_dim1 hi | tile_dim0
    g1[4] = 32u;                                          // tile_dim1 (tile_dim2=0)
    g1[5] = (unsigned)V_;                                 // tensor_dim0_stride[31:0]
    g1[6] = 0u;                                           // stride hi | dim1_stride lo
    g1[7] = 0u;
    asm volatile("tensor_load_to_lds %0, %1" :: "s"(g0), "s"(g1) : "memory");
}

// ---- small utility kernels -------------------------------------------------
__global__ void f32_to_bf16_kernel(const float* __restrict__ s, __bf16* __restrict__ d, int n) {
    int i = blockIdx.x * blockDim.x + threadIdx.x;
    if (i < n) d[i] = (__bf16)s[i];
}

__global__ void fzero_kernel(float* __restrict__ p, int n) {
    int i = blockIdx.x * blockDim.x + threadIdx.x;
    if (i < n) p[i] = 0.0f;
}

__global__ void x0_broadcast_kernel(const float* __restrict__ ie,
                                    const float* __restrict__ g,
                                    const float* __restrict__ b,
                                    __bf16* __restrict__ xseq0) {
    int idx = blockIdx.x * blockDim.x + threadIdx.x;   // over B_*E_
    int e   = idx & (E_ - 1);
    float s = 0.f, q = 0.f;
#pragma unroll
    for (int j = 0; j < E_; ++j) { float v = ie[j]; s += v; q += v * v; }
    float mean = s * (1.0f / E_);
    float rstd = rsqrtf(q * (1.0f / E_) - mean * mean + LN_EPS);
    xseq0[idx] = (__bf16)((ie[e] - mean) * rstd * g[e] + b[e]);
}

// ---- embedding GEMM + bias + LN(E), TDM-staged B panel ---------------------
// grid (B_/128, 1, T_-1); block 256 = 8 waves; wave -> 16-row M tile, N=0..31.
__global__ void embed_gemm_ln_kernel(const float* __restrict__ msg,
                                     const __bf16* __restrict__ Wemb,   // col-major [E_][V_]
                                     const float* __restrict__ eb,
                                     const float* __restrict__ g,
                                     const float* __restrict__ bb,
                                     __bf16* __restrict__ xseq) {
    __shared__ __bf16 Bs[2][32 * CK_];    // double-buffered B panel (32 KB)
    const int lane = threadIdx.x & 31;
    const int w    = threadIdx.x >> 5;
    const int t    = blockIdx.z;
    const int m0   = blockIdx.x * 128 + w * 16;
    const int mrow = m0 + (lane & 15);
    const float* Arow = msg + ((size_t)mrow * T_ + t) * V_;

    if (w == 0) {                          // wave 0 drives the TDM pipeline
        tdm_load_panel(Bs[0], Wemb);
        wait_tensorcnt0();
    }
    __syncthreads();

    v8f acc0 = {}; v8f acc1 = {};
    for (int c = 0; c < V_ / CK_; ++c) {
        const int k0 = c * CK_;
        if (w == 0 && c + 1 < V_ / CK_)    // prefetch next chunk into other buffer
            tdm_load_panel(Bs[(c + 1) & 1], Wemb + (size_t)(k0 + CK_));
        const __bf16* Bp = Bs[c & 1];
#pragma unroll 2
        for (int kk = 0; kk < CK_; kk += 32) {
            __builtin_prefetch(Arow + k0 + kk + 128, 0, 0);
            v16bf a  = load_a_from_f32(Arow, k0 + kk, lane);
            v16bf b0 = load_b_panel(Bp, CK_, (lane & 15),      kk, lane);
            v16bf b1 = load_b_panel(Bp, CK_, 16 + (lane & 15), kk, lane);
            acc0 = WMMA_BF16(a, b0, acc0);
            acc1 = WMMA_BF16(a, b1, acc1);
        }
        if (w == 0) wait_tensorcnt0();
        __syncthreads();
    }

    const int nc0 = lane & 15, nc1 = 16 + (lane & 15);
    const float be0 = eb[nc0], be1 = eb[nc1];
    const float g0 = g[nc0], g1 = g[nc1], bb0 = bb[nc0], bb1 = bb[nc1];
#pragma unroll
    for (int i = 0; i < 8; ++i) { acc0[i] += be0; acc1[i] += be1; }

    // LN over E=32: row values live across this lane's 16-lane half in acc0/acc1
#pragma unroll
    for (int i = 0; i < 8; ++i) {
        float s = acc0[i] + acc1[i];
        float q = acc0[i] * acc0[i] + acc1[i] * acc1[i];
        SWZ_ADD(s, 0x41f);  SWZ_ADD(q, 0x41f);    // xor 1
        SWZ_ADD(s, 0x81f);  SWZ_ADD(q, 0x81f);    // xor 2
        SWZ_ADD(s, 0x101f); SWZ_ADD(q, 0x101f);   // xor 4
        SWZ_ADD(s, 0x201f); SWZ_ADD(q, 0x201f);   // xor 8
        float mean = s * (1.0f / 32.0f);
        float rstd = rsqrtf(q * (1.0f / 32.0f) - mean * mean + LN_EPS);
        int   M    = i + ((lane >> 4) << 3);
        size_t ro  = ((size_t)(t + 1) * B_ + (m0 + M)) * E_;
        xseq[ro + nc0] = (__bf16)((acc0[i] - mean) * rstd * g0 + bb0);
        xseq[ro + nc1] = (__bf16)((acc1[i] - mean) * rstd * g1 + bb1);
    }
}

// ---- fused GRU step with LDS-shared A strips -------------------------------
// grid (B_/16, 2); block 256 = 8 waves. Block owns one 16-row M strip; the h/x
// strips are staged once in LDS (bf16) and shared by all waves. Each wave
// computes 2 adjacent 16-col tiles of h_new (block covers 256 of 512 cols).
__global__ void gru_step_kernel(const __bf16* __restrict__ x,     // [B_, E_]
                                const float*  __restrict__ h,     // [B_, H_]
                                const __bf16* __restrict__ WihT,  // col-major [H3_][E_]
                                const __bf16* __restrict__ WhhT,  // col-major [H3_][H_]
                                const float*  __restrict__ bih,
                                const float*  __restrict__ bhh,
                                float* __restrict__ hnew) {
    __shared__ __bf16 hs[16 * H_];   // 16 KB
    __shared__ __bf16 xs[16 * E_];   // 1 KB
    const int lane = threadIdx.x & 31;
    const int w    = threadIdx.x >> 5;
    const int tid  = threadIdx.x;
    const int m0   = blockIdx.x * 16;

    for (int i = tid; i < 16 * H_; i += 256) {
        int r = i >> 9, c = i & (H_ - 1);
        hs[i] = (__bf16)h[(size_t)(m0 + r) * H_ + c];
    }
    for (int i = tid; i < 16 * E_; i += 256)
        xs[i] = x[(size_t)m0 * E_ + i];
    __syncthreads();

    const int n0  = blockIdx.y * 256 + w * 32;
    const int nc0 = n0 + (lane & 15);
    const int nc1 = n0 + 16 + (lane & 15);
    const __bf16* arow = hs + (size_t)(lane & 15) * H_;

    v8f aR0 = {}, aZ0 = {}, aNi0 = {}, aNh0 = {};
    v8f aR1 = {}, aZ1 = {}, aNi1 = {}, aNh1 = {};

    // input-hidden: K = E_ = 32 -> one WMMA per gate per tile
    {
        v16bf ax = load_a_from_bf16(xs + (size_t)(lane & 15) * E_, 0, lane);
        aR0  = WMMA_BF16(ax, load_b_panel(WihT, E_, nc0,        0, lane), aR0);
        aZ0  = WMMA_BF16(ax, load_b_panel(WihT, E_, nc0 + 512,  0, lane), aZ0);
        aNi0 = WMMA_BF16(ax, load_b_panel(WihT, E_, nc0 + 1024, 0, lane), aNi0);
        aR1  = WMMA_BF16(ax, load_b_panel(WihT, E_, nc1,        0, lane), aR1);
        aZ1  = WMMA_BF16(ax, load_b_panel(WihT, E_, nc1 + 512,  0, lane), aZ1);
        aNi1 = WMMA_BF16(ax, load_b_panel(WihT, E_, nc1 + 1024, 0, lane), aNi1);
    }

    // hidden-hidden: K = H_ = 512, A re-used from LDS across both tiles
    for (int k0 = 0; k0 < H_; k0 += 32) {
        v16bf ah = load_a_from_bf16(arow, k0, lane);
        aR0  = WMMA_BF16(ah, load_b_panel(WhhT, H_, nc0,        k0, lane), aR0);
        aZ0  = WMMA_BF16(ah, load_b_panel(WhhT, H_, nc0 + 512,  k0, lane), aZ0);
        aNh0 = WMMA_BF16(ah, load_b_panel(WhhT, H_, nc0 + 1024, k0, lane), aNh0);
        aR1  = WMMA_BF16(ah, load_b_panel(WhhT, H_, nc1,        k0, lane), aR1);
        aZ1  = WMMA_BF16(ah, load_b_panel(WhhT, H_, nc1 + 512,  k0, lane), aZ1);
        aNh1 = WMMA_BF16(ah, load_b_panel(WhhT, H_, nc1 + 1024, k0, lane), aNh1);
    }

#pragma unroll
    for (int tt = 0; tt < 2; ++tt) {
        const int nc = tt ? nc1 : nc0;
        v8f aR  = tt ? aR1  : aR0;
        v8f aZ  = tt ? aZ1  : aZ0;
        v8f aNi = tt ? aNi1 : aNi0;
        v8f aNh = tt ? aNh1 : aNh0;
        const float brC  = bih[nc]        + bhh[nc];
        const float bzC  = bih[nc + 512]  + bhh[nc + 512];
        const float bniC = bih[nc + 1024];
        const float bnhC = bhh[nc + 1024];
#pragma unroll
        for (int i = 0; i < 8; ++i) {
            int   M  = i + ((lane >> 4) << 3);
            float r  = sigmoidf_(aR[i] + brC);
            float z  = sigmoidf_(aZ[i] + bzC);
            float n  = tanhf(aNi[i] + bniC + r * (aNh[i] + bnhC));
            size_t o = (size_t)(m0 + M) * H_ + nc;
            float hp = h[o];                 // fp32 previous state (exact)
            hnew[o]  = (1.0f - z) * n + z * hp;
        }
    }
}

// ---- fused LN(H) + FC + sigmoid --------------------------------------------
// grid (B_/16, 2); block 256 = 8 waves. LN-ed activations staged bf16 in LDS,
// each wave computes 4 N-tiles (block covers 512 of 1024 output cols).
__global__ void ln_fc_sigmoid_kernel(const float* __restrict__ h,
                                     const float* __restrict__ g,
                                     const float* __restrict__ b,
                                     const __bf16* __restrict__ WT,   // col-major [O_][H_]
                                     const float* __restrict__ bias,
                                     float* __restrict__ out) {
    __shared__ __bf16 as[16 * H_];   // 16 KB
    const int lane = threadIdx.x & 31;
    const int w    = threadIdx.x >> 5;
    const int m0   = blockIdx.x * 16;

    // LN: each wave normalizes 2 rows of the strip
#pragma unroll
    for (int rr = 0; rr < 2; ++rr) {
        const int r = w * 2 + rr;
        const float* hr = h + (size_t)(m0 + r) * H_;
        float s = 0.f, q = 0.f;
#pragma unroll
        for (int j = lane; j < H_; j += 32) { float v = hr[j]; s += v; q += v * v; }
        SWZ_ADD(s, 0x41f);  SWZ_ADD(q, 0x41f);
        SWZ_ADD(s, 0x81f);  SWZ_ADD(q, 0x81f);
        SWZ_ADD(s, 0x101f); SWZ_ADD(q, 0x101f);
        SWZ_ADD(s, 0x201f); SWZ_ADD(q, 0x201f);
        SWZ_ADD(s, 0x401f); SWZ_ADD(q, 0x401f);   // xor 16 -> full 32-lane sum
        float mean = s * (1.0f / H_);
        float rstd = rsqrtf(q * (1.0f / H_) - mean * mean + LN_EPS);
#pragma unroll
        for (int j = lane; j < H_; j += 32)
            as[r * H_ + j] = (__bf16)((hr[j] - mean) * rstd * g[j] + b[j]);
    }
    __syncthreads();

    const int nb = blockIdx.y * 512 + w * 64;
    const __bf16* arow = as + (size_t)(lane & 15) * H_;
    v8f acc[4] = {};
    for (int k0 = 0; k0 < H_; k0 += 32) {
        v16bf a = load_a_from_bf16(arow, k0, lane);
#pragma unroll
        for (int tt = 0; tt < 4; ++tt) {
            v16bf bm = load_b_panel(WT, H_, nb + tt * 16 + (lane & 15), k0, lane);
            acc[tt] = WMMA_BF16(a, bm, acc[tt]);
        }
    }
#pragma unroll
    for (int tt = 0; tt < 4; ++tt) {
        const int ncol = nb + tt * 16 + (lane & 15);
        const float bc = bias[ncol];
#pragma unroll
        for (int i = 0; i < 8; ++i) {
            int M = i + ((lane >> 4) << 3);
            out[(size_t)(m0 + M) * O_ + ncol] = sigmoidf_(acc[tt][i] + bc);
        }
    }
}

// ---- host-side launch ------------------------------------------------------
extern "C" void kernel_launch(void* const* d_in, const int* in_sizes, int n_in,
                              void* d_out, int out_size, void* d_ws, size_t ws_size,
                              hipStream_t stream) {
    const float* message  = (const float*)d_in[0];
    const float* emb_W    = (const float*)d_in[1];
    const float* emb_b    = (const float*)d_in[2];
    const float* init_emb = (const float*)d_in[3];
    const float* eln_g    = (const float*)d_in[4];
    const float* eln_b    = (const float*)d_in[5];
    const float* Wih      = (const float*)d_in[6];
    const float* Whh      = (const float*)d_in[7];
    const float* bih      = (const float*)d_in[8];
    const float* bhh      = (const float*)d_in[9];
    const float* gln_g    = (const float*)d_in[10];
    const float* gln_b    = (const float*)d_in[11];
    const float* fc_W     = (const float*)d_in[12];
    const float* fc_b     = (const float*)d_in[13];
    float* out = (float*)d_out;

    char* ws = (char*)d_ws;
    size_t off = 0;
    auto carve = [&](size_t bytes) -> char* {
        char* p = ws + off;
        off = (off + bytes + 255) & ~(size_t)255;
        return p;
    };
    __bf16* embWb = (__bf16*)carve((size_t)V_ * E_ * 2);
    __bf16* WihTb = (__bf16*)carve((size_t)H3_ * E_ * 2);
    __bf16* WhhTb = (__bf16*)carve((size_t)H3_ * H_ * 2);
    __bf16* fcWTb = (__bf16*)carve((size_t)O_ * H_ * 2);
    __bf16* xseq  = (__bf16*)carve((size_t)T_ * B_ * E_ * 2);
    float*  h0    = (float*) carve((size_t)B_ * H_ * 4);
    float*  h1    = (float*) carve((size_t)B_ * H_ * 4);

    // 1) weight panels fp32 -> bf16 (already B-panel friendly layouts)
    f32_to_bf16_kernel<<<(V_ * E_ + 255) / 256, 256, 0, stream>>>(emb_W, embWb, V_ * E_);
    f32_to_bf16_kernel<<<(H3_ * E_ + 255) / 256, 256, 0, stream>>>(Wih, WihTb, H3_ * E_);
    f32_to_bf16_kernel<<<(H3_ * H_ + 255) / 256, 256, 0, stream>>>(Whh, WhhTb, H3_ * H_);
    f32_to_bf16_kernel<<<(O_ * H_ + 255) / 256, 256, 0, stream>>>(fc_W, fcWTb, O_ * H_);

    // 2) x0 = LN(init_emb) broadcast into xseq slot 0
    x0_broadcast_kernel<<<(B_ * E_) / 256, 256, 0, stream>>>(init_emb, eln_g, eln_b, xseq);

    // 3) embedding GEMM (TDM-staged B) + bias + LN into xseq slots 1..T-1
    embed_gemm_ln_kernel<<<dim3(B_ / 128, 1, T_ - 1), 256, 0, stream>>>(
        message, embWb, emb_b, eln_g, eln_b, xseq);

    // 4) h = 0, then 5 fused GRU steps (ping-pong)
    fzero_kernel<<<(B_ * H_ + 255) / 256, 256, 0, stream>>>(h0, B_ * H_);
    float* hc = h0;
    float* hn = h1;
    for (int t = 0; t < T_; ++t) {
        gru_step_kernel<<<dim3(B_ / 16, 2), 256, 0, stream>>>(
            xseq + (size_t)t * B_ * E_, hc, WihTb, WhhTb, bih, bhh, hn);
        float* tmp = hc; hc = hn; hn = tmp;
    }

    // 5) fused LN(H) + FC + sigmoid into d_out
    ln_fc_sigmoid_kernel<<<dim3(B_ / 16, 2), 256, 0, stream>>>(
        hc, gln_g, gln_b, fcWTb, fc_b, out);

    (void)in_sizes; (void)n_in; (void)out_size; (void)ws_size;
}
